// QSelfAttention_48859547959351
// MI455X (gfx1250) — matrix operand order repaired and checked by reading, not compiled, not run on hardware
//
#include <hip/hip_runtime.h>
#include <hip/hip_bf16.h>

typedef __attribute__((ext_vector_type(16))) _Float16 v16h;
typedef __attribute__((ext_vector_type(8)))  float    v8f;

#define QN 8
#define DIM 256
#define NB 8
#define NQ 32
#define NK 32

__device__ __forceinline__ float2 cmulf(float2 a, float2 b) {
  return make_float2(a.x*b.x - a.y*b.y, a.x*b.y + a.y*b.x);
}
__device__ __forceinline__ float2 caddf(float2 a, float2 b) {
  return make_float2(a.x + b.x, a.y + b.y);
}

// U = RZ(c) * RY(b) * RZ(a)   (RZ(a) applied first)
__device__ __forceinline__ void zyz_u(float a, float b, float c,
                                      float2& u00, float2& u01, float2& u10, float2& u11) {
  float cb = cosf(0.5f*b), sb = sinf(0.5f*b);
  float p  = 0.5f*(a + c), m = 0.5f*(a - c);
  float cp = cosf(p), sp = sinf(p);
  float cm = cosf(m), sm = sinf(m);
  u00 = make_float2( cb*cp, -cb*sp);   //  cb * e^{-i(a+c)/2}
  u01 = make_float2(-sb*cm, -sb*sm);   // -sb * e^{+i(a-c)/2}
  u10 = make_float2( sb*cm, -sb*sm);   //  sb * e^{-i(a-c)/2}
  u11 = make_float2( cb*cp,  cb*sp);   //  cb * e^{+i(a+c)/2}
}

// Apply 2x2 complex U on qubit w (qubit 0 = MSB) of an N-qubit LDS statevector.
template<int N, int NT>
__device__ __forceinline__ void apply_u(float2* st, int w,
                                        float2 u00, float2 u01, float2 u10, float2 u11,
                                        int tid) {
  const int pairs = 1 << (N - 1);
  const int s = 1 << (N - 1 - w);
  __syncthreads();
  for (int pi = tid; pi < pairs; pi += NT) {
    int i0 = ((pi & ~(s - 1)) << 1) | (pi & (s - 1));
    int i1 = i0 + s;
    float2 x0 = st[i0], x1 = st[i1];
    st[i0] = caddf(cmulf(u00, x0), cmulf(u01, x1));
    st[i1] = caddf(cmulf(u10, x0), cmulf(u11, x1));
  }
}

template<int N, int NT>
__device__ __forceinline__ void apply_cnot_g(float2* st, int c, int t, int tid) {
  const int pairs = 1 << (N - 1);
  const int s  = 1 << (N - 1 - t);
  const int cb = 1 << (N - 1 - c);
  __syncthreads();
  for (int pi = tid; pi < pairs; pi += NT) {
    int i0 = ((pi & ~(s - 1)) << 1) | (pi & (s - 1));
    if (i0 & cb) {
      int i1 = i0 + s;
      float2 tmp = st[i0]; st[i0] = st[i1]; st[i1] = tmp;
    }
  }
}

// ---------------- Kernel 1: v_circuit (256 independent 8-qubit sims) ----------------
__global__ void __launch_bounds__(128)
qa_vcirc(const float* __restrict__ value, const float* __restrict__ vp,
         float* __restrict__ vr, float* __restrict__ vi) {
  __shared__ float2 st[256];
  const int vidx = blockIdx.x;          // b*NK + k
  const int tid = threadIdx.x;
  const float* vrow = value + vidx * QN;
  for (int i = tid; i < 256; i += 128) st[i] = make_float2(i == 0 ? 1.f : 0.f, 0.f);
  float2 u00, u01, u10, u11;
  for (int i = 0; i < QN; ++i) {
    zyz_u(0.f, vrow[i], 0.f, u00, u01, u10, u11);
    apply_u<8, 128>(st, i, u00, u01, u10, u11, tid);
  }
  for (int rep = 0; rep < 6; ++rep) {
    for (int i = 0; i < QN; ++i) {
      float a = vp[(rep*3 + 0)*QN + i];
      float b = vp[(rep*3 + 1)*QN + i];
      float c = vp[(rep*3 + 2)*QN + i];
      zyz_u(a, b, c, u00, u01, u10, u11);
      apply_u<8, 128>(st, i, u00, u01, u10, u11, tid);
    }
    for (int i = 0; i < QN; ++i) {
      int tg = (i + 1 + rep) % QN;
      if (tg == i) tg = (tg + 1) % QN;
      apply_cnot_g<8, 128>(st, i, tg, tid);
    }
  }
  __syncthreads();
  for (int d = tid; d < 256; d += 128) {
    vr[vidx*256 + d] = st[d].x;
    vi[vidx*256 + d] = st[d].y;
  }
}

// ---------------- Kernel 2: attn pairs (8192 independent 9-qubit sims) ----------------
__global__ void __launch_bounds__(256)
qa_pairs(const float* __restrict__ query, const float* __restrict__ key,
         const float* __restrict__ qkp, const float* __restrict__ ap,
         float* __restrict__ a0r, float* __restrict__ a0i,
         float* __restrict__ a1r, float* __restrict__ a1i) {
  __shared__ float2 st[512];
  const int p = blockIdx.x;             // b*1024 + q*32 + k
  const int tid = threadIdx.x;
  const int b = p >> 10, q = (p >> 5) & 31, k = p & 31;
  const float* qrow = query + (b*NQ + q)*QN;
  const float* krow = key   + (b*NK + k)*QN;
  for (int i = tid; i < 512; i += 256) st[i] = make_float2(i == 0 ? 1.f : 0.f, 0.f);
  float2 u00, u01, u10, u11;
  for (int i = 0; i < QN; ++i) {                              // RY(q_i) on qubit i+1
    zyz_u(0.f, qrow[i], 0.f, u00, u01, u10, u11);
    apply_u<9, 256>(st, i + 1, u00, u01, u10, u11, tid);
  }
  for (int i = 0; i < QN; ++i) {                              // RY(qk1)·RZ(qk0)
    zyz_u(qkp[i], qkp[QN + i], 0.f, u00, u01, u10, u11);
    apply_u<9, 256>(st, i + 1, u00, u01, u10, u11, tid);
  }
  for (int i = 0; i < QN; ++i) {                              // RY(-k_i)
    zyz_u(0.f, -krow[i], 0.f, u00, u01, u10, u11);
    apply_u<9, 256>(st, i + 1, u00, u01, u10, u11, tid);
  }
  // X on qubit 0
  apply_u<9, 256>(st, 0, make_float2(0.f,0.f), make_float2(1.f,0.f),
                         make_float2(1.f,0.f), make_float2(0.f,0.f), tid);
  for (int i = 0; i < QN - 1; ++i) {
    zyz_u(0.f, ap[i], 0.f, u00, u01, u10, u11);
    apply_u<9, 256>(st, i + 1, u00, u01, u10, u11, tid);
    zyz_u(0.f, ap[QN + i], 0.f, u00, u01, u10, u11);
    apply_u<9, 256>(st, i + 2, u00, u01, u10, u11, tid);
    apply_cnot_g<9, 256>(st, i + 1, i + 2, tid);
  }
  zyz_u(0.f, ap[QN - 1], 0.f, u00, u01, u10, u11);
  apply_u<9, 256>(st, QN, u00, u01, u10, u11, tid);
  zyz_u(0.f, ap[2*QN - 1], 0.f, u00, u01, u10, u11);
  apply_u<9, 256>(st, 0, u00, u01, u10, u11, tid);
  apply_cnot_g<9, 256>(st, QN, 0, tid);
  __syncthreads();
  if (tid == 0) {
    a0r[p] = st[0].x;   a0i[p] = st[0].y;
    a1r[p] = st[256].x; a1i[p] = st[256].y;
  }
}

// ---------------- Kernel 3: suffix-product weights ----------------
__global__ void __launch_bounds__(256)
qa_weights(const float* __restrict__ a0r, const float* __restrict__ a0i,
           const float* __restrict__ a1r, const float* __restrict__ a1i,
           float* __restrict__ wr, float* __restrict__ wi,
           float* __restrict__ totr, float* __restrict__ toti) {
  const int bq = threadIdx.x;           // 0..255
  float sr = 1.f, si = 0.f;             // suffix = prod_{j>k} a0[j]
  for (int k = NK - 1; k >= 0; --k) {
    int idx = bq*NK + k;
    float ar = a1r[idx], ai = a1i[idx];
    wr[idx] = sr*ar - si*ai;
    wi[idx] = sr*ai + si*ar;
    float br = a0r[idx], bi = a0i[idx];
    float nsr = sr*br - si*bi;
    si = sr*bi + si*br;
    sr = nsr;
  }
  totr[bq] = sr; toti[bq] = si;         // total = prod_k a0[k]
}

// ---------------- Kernel 4: complex batched GEMM via WMMA (K = 32 = one f16 WMMA step) ----------------
#define WMMA_F16(A, Bm, C) \
  __builtin_amdgcn_wmma_f32_16x16x32_f16(false, (A), false, (Bm), (short)0, (C), false, false)

__global__ void __launch_bounds__(32)
qa_gemm(const float* __restrict__ wr, const float* __restrict__ wi,
        const float* __restrict__ vr, const float* __restrict__ vi,
        float* __restrict__ vnr, float* __restrict__ vni) {
  const int tile = blockIdx.x;          // 256 tiles: b(8) x mt(2) x nt(16)
  const int nt = tile & 15;
  const int mt = (tile >> 4) & 1;
  const int b  = tile >> 5;
  const int lane = threadIdx.x;
  const int lo = lane & 15, half = lane >> 4;

  const float* wrb = wr + b*NQ*NK;
  const float* wib = wi + b*NQ*NK;
  const float* vrb = vr + b*NK*DIM;
  const float* vib = vi + b*NK*DIM;

  v16h a_rh = {}, a_rl = {}, a_ih = {}, a_il = {};
  v16h b_rh = {}, b_rl = {}, b_ih = {}, b_il = {};

  // A tile (16x32): lane holds row M = lo, slot j -> K = (j&7) + ((j&8)<<1) + 8*half
  const int arow = (mt*16 + lo) * NK;
#pragma unroll
  for (int j = 0; j < 16; ++j) {
    int kk = (j & 7) + ((j & 8) << 1) + (half << 3);
    float xr = wrb[arow + kk];
    _Float16 hr = (_Float16)xr; a_rh[j] = hr; a_rl[j] = (_Float16)(xr - (float)hr);
    float xi = wib[arow + kk];
    _Float16 hi = (_Float16)xi; a_ih[j] = hi; a_il[j] = (_Float16)(xi - (float)hi);
  }
  // B tile (32x16): lane holds col N = lo, slot j -> K = (j&7) + ((j&8)<<1) + 8*half
  const int bcol = nt*16 + lo;
#pragma unroll
  for (int j = 0; j < 16; ++j) {
    int kk = (j & 7) + ((j & 8) << 1) + (half << 3);
    float xr = vrb[kk*DIM + bcol];
    _Float16 hr = (_Float16)xr; b_rh[j] = hr; b_rl[j] = (_Float16)(xr - (float)hr);
    float xi = vib[kk*DIM + bcol];
    _Float16 hi = (_Float16)xi; b_ih[j] = hi; b_il[j] = (_Float16)(xi - (float)hi);
  }
  v16h nb_ih = -b_ih, nb_il = -b_il;    // f16 NEG not allowed on A/B for WMMA: negate in VALU

  // real = wr*vr - wi*vi   (hi*hi + hi*lo + lo*hi split)
  v8f cr = {};
  cr = WMMA_F16(a_rh, b_rh, cr);
  cr = WMMA_F16(a_rh, b_rl, cr);
  cr = WMMA_F16(a_rl, b_rh, cr);
  cr = WMMA_F16(a_ih, nb_ih, cr);
  cr = WMMA_F16(a_ih, nb_il, cr);
  cr = WMMA_F16(a_il, nb_ih, cr);
  // imag = wr*vi + wi*vr
  v8f ci = {};
  ci = WMMA_F16(a_rh, b_ih, ci);
  ci = WMMA_F16(a_rh, b_il, ci);
  ci = WMMA_F16(a_rl, b_ih, ci);
  ci = WMMA_F16(a_ih, b_rh, ci);
  ci = WMMA_F16(a_ih, b_rl, ci);
  ci = WMMA_F16(a_il, b_rh, ci);

  // C tile: lane L, VGPR r -> M = r + 8*half, N = lo
  float* onr = vnr + b*NQ*DIM;
  float* oni = vni + b*NQ*DIM;
#pragma unroll
  for (int r = 0; r < 8; ++r) {
    int qrow = mt*16 + r + half*8;
    onr[qrow*DIM + bcol] = cr[r];
    oni[qrow*DIM + bcol] = ci[r];
  }
}

// ---------------- Kernel 5: probs + signed bit-parity reduction ----------------
__global__ void __launch_bounds__(256)
qa_final(const float* __restrict__ vnr, const float* __restrict__ vni,
         const float* __restrict__ totr, const float* __restrict__ toti,
         float* __restrict__ out) {
  const int bq = threadIdx.x;           // 0..255
  const float* rr = vnr + bq*DIM;
  const float* ri = vni + bq*DIM;
  float S = 0.f, Sw[QN] = {0.f,0.f,0.f,0.f,0.f,0.f,0.f,0.f};
  for (int d = 0; d < DIM; ++d) {
    float re = rr[d] + (d == 0 ? totr[bq] : 0.f);
    float im = ri[d] + (d == 0 ? toti[bq] : 0.f);
    float pr = re*re + im*im;
    S += pr;
#pragma unroll
    for (int w = 0; w < QN; ++w)
      if ((d >> (QN - 1 - w)) & 1) Sw[w] += pr;
  }
#pragma unroll
  for (int w = 0; w < QN; ++w)
    out[bq*QN + w] = (S - 2.f*Sw[w]) / S;
}

extern "C" void kernel_launch(void* const* d_in, const int* in_sizes, int n_in,
                              void* d_out, int out_size, void* d_ws, size_t ws_size,
                              hipStream_t stream) {
  (void)in_sizes; (void)n_in; (void)out_size; (void)ws_size;
  const float* query = (const float*)d_in[0];   // (8,32,8)
  const float* key   = (const float*)d_in[1];   // (8,32,8)
  const float* value = (const float*)d_in[2];   // (8,32,8)
  const float* vpar  = (const float*)d_in[3];   // (6,3,8)
  const float* qkp   = (const float*)d_in[4];   // (3,8)
  const float* ap    = (const float*)d_in[5];   // (2,8)
  float* out = (float*)d_out;                   // (8,32,8)

  float* ws = (float*)d_ws;
  float* vr   = ws;            // 65536  v_states real (B*K, 256)
  float* vi   = ws + 65536;    // 65536  v_states imag
  float* a0r  = ws + 131072;   // 8192
  float* a0i  = ws + 139264;   // 8192
  float* a1r  = ws + 147456;   // 8192
  float* a1i  = ws + 155648;   // 8192
  float* wr   = ws + 163840;   // 8192   weights real
  float* wi   = ws + 172032;   // 8192   weights imag
  float* totr = ws + 180224;   // 256
  float* toti = ws + 180480;   // 256
  float* vnr  = ws + 180736;   // 65536  v_new real
  float* vni  = ws + 246272;   // 65536  v_new imag

  qa_vcirc<<<NB*NK, 128, 0, stream>>>(value, vpar, vr, vi);
  qa_pairs<<<NB*NQ*NK, 256, 0, stream>>>(query, key, qkp, ap, a0r, a0i, a1r, a1i);
  qa_weights<<<1, 256, 0, stream>>>(a0r, a0i, a1r, a1i, wr, wi, totr, toti);
  qa_gemm<<<NB*2*16, 32, 0, stream>>>(wr, wi, vr, vi, vnr, vni);
  qa_final<<<1, 256, 0, stream>>>(vnr, vni, totr, toti, out);
}
